// edc_loss_23691039605308
// MI455X (gfx1250) — matrix operand order repaired and checked by reading, not compiled, not run on hardware
//
#include <hip/hip_runtime.h>
#include <hip/hip_bf16.h>
#include <math.h>

// ---------------------------------------------------------------------------
// EDC loss on MI455X (gfx1250, wave32, WMMA)
//   Kernel P: pack A = scaled re/im spectra -> bf16, pre-swizzled in WMMA
//             fragment order (one-time, 12.3 MB, L2-resident).
//   Kernel 1: irfft as DFT GEMM on the bf16 WMMA pipe. Twiddles generated by
//             register-resident complex rotation recurrence (no per-iter trig).
//   Kernel 3: 9-band x 7-section DF2T biquad cascade + Schroeder |suffix| loss.
// Workspace: rir[2][47040][64] f32 (24.1 MB) + A_pre (12.3 MB) = 36.4 MB.
// ---------------------------------------------------------------------------

typedef __attribute__((ext_vector_type(16))) __bf16 v16bf;
typedef __attribute__((ext_vector_type(8)))  float  v8f;

#define SRN    48000
#define NFREQ  24001          // rfft bins used by irfft(n=48000)
#define KTOT   48002          // 2*NFREQ (re/im interleaved along K)
#define MIX    960
#define T_OUT  47040          // 48000 - 960
#define BATCH  64
#define NSEC   7
#define NBAND  9
#define KTILE  32
#define KTILES 1501           // ceil(KTOT/KTILE)
#define NWG    128            // output columns per workgroup
#define BSTR   136            // padded LDS row stride for B tile (272B, 16B-aligned)
#define APRE_PER_SIG ((size_t)KTILES * 2048)   // 2048 bf16 per k-tile

union Oct { uint4 q; __bf16 h[8]; };

// ---------------------------------------------------------------------------
// Kernel P: A_pre[sig][kt][half][m][e] bf16, e in fragment order:
//   kk = ((e&8)<<1) + half*8 + (e&7);  kp = kt*32+kk;  k = kp>>1
//   value = c_k/N * (kp odd ? Im : Re)[m][k], zero-padded (incl. DC/Nyq imag)
// ---------------------------------------------------------------------------
__global__ __launch_bounds__(256)
void pack_a_bf16(const float* __restrict__ tre, const float* __restrict__ tim,
                 const float* __restrict__ are, const float* __restrict__ aim,
                 __bf16* __restrict__ apre) {
    size_t g = (size_t)blockIdx.x * 256 + threadIdx.x;  // < 2*KTILES*2048
    int e  = (int)(g & 15);
    int m  = (int)((g >> 4) & 63);
    int h  = (int)((g >> 10) & 1);
    size_t r = g >> 11;
    int kt  = (int)(r % KTILES);
    int sig = (int)(r / KTILES);
    const float* __restrict__ re = sig ? are : tre;
    const float* __restrict__ im = sig ? aim : tim;

    int kk = ((e & 8) << 1) + h * 8 + (e & 7);
    int kp = kt * KTILE + kk;
    float v = 0.0f;
    if (kp < KTOT && kp != 1 && kp != KTOT - 1) {
        int k = kp >> 1;
        float sc = (k == 0 || k == NFREQ - 1) ? (1.0f / (float)SRN)
                                              : (2.0f / (float)SRN);
        v = ((kp & 1) ? im[m * SRN + k] : re[m * SRN + k]) * sc;
    }
    apre[g] = (__bf16)v;
}

// ---------------------------------------------------------------------------
// Kernel 1: DFT-as-GEMM via v_wmma_f32_16x16x32_bf16.
// Block = 256 threads (8 waves); per-WG tile 64(M) x 128(N); K loop over 1501
// tiles of 32 (re/im-interleaved K). A fragments load straight from global
// (pre-swizzled, L2-resident). B tile staged in LDS by complex rotation:
// thread (col, kh) owns column t=MIX+n0+col and k-offsets kh*8..kh*8+7;
// per k-step rotate by e^{i*2pi*t/N}; per tile jump 8 more; renormalize.
// ---------------------------------------------------------------------------
__global__ __launch_bounds__(256)
void dft_irfft_wmma(const __bf16* __restrict__ apre_all,
                    float* __restrict__ rir /* [2][T_OUT][BATCH] */) {
    const int sig = blockIdx.y;
    const __bf16* __restrict__ apre = apre_all + (size_t)sig * APRE_PER_SIG;
    float* __restrict__ out = rir + (size_t)sig * (size_t)T_OUT * BATCH;

    __shared__ __attribute__((aligned(16))) __bf16 Bs[KTILE * BSTR];

    const int tid  = threadIdx.x;
    const int lane = tid & 31;
    const int w    = tid >> 5;
    const int mt   = w & 3;            // M tile (rows mt*16..+15)
    const int jg   = w >> 2;           // N group (0/1): columns jg*64..+63
    const int half = lane >> 4;
    const int mrow = (mt << 4) + (lane & 15);
    const int n0   = blockIdx.x * NWG;

    const float wrad = 6.28318530717958647692f / (float)SRN;  // 2*pi/48000

    // --- B-staging rotation state (register resident across whole K loop) ---
    const int col = tid & 127;
    const int kh  = tid >> 7;                  // 0/1: k-offsets kh*8..kh*8+7
    const unsigned t = (unsigned)(MIX + n0 + col);
    float c1, s1, c8, s8, c, s;
    __sincosf(wrad * (float)(t % (unsigned)SRN), &s1, &c1);            // e^{i*th*t}
    __sincosf(wrad * (float)((8u * t) % (unsigned)SRN), &s8, &c8);     // e^{i*8*th*t}
    __sincosf(wrad * (float)(((unsigned)(kh * 8) * t) % (unsigned)SRN), &s, &c);

    v8f acc[4] = {};

    for (int kt = 0; kt < KTILES; ++kt) {
        // ---- A fragment: two contiguous b128 loads from pre-swizzled global
        const uint4* ap = (const uint4*)(apre + (size_t)kt * 2048
                                              + (size_t)(half * 64 + mrow) * 16);
        Oct a0, a1;
        a0.q = ap[0];
        a1.q = ap[1];

        // ---- stage B tile [32 kp][128 nn] via rotation recurrence
        float cc = c, ss = s;
        #pragma unroll
        for (int st = 0; st < 8; ++st) {
            int kk = (kh * 8 + st) * 2;
            Bs[kk * BSTR + col]       = (__bf16)cc;        // cos row
            Bs[(kk + 1) * BSTR + col] = (__bf16)(-ss);     // -sin row
            float cn = fmaf(cc, c1, -ss * s1);
            ss = fmaf(ss, c1, cc * s1);
            cc = cn;
        }
        // jump over the partner thread's 8 k's, then renormalize drift
        {
            float cn = fmaf(cc, c8, -ss * s8);
            ss = fmaf(ss, c8, cc * s8);
            cc = cn;
            float inv = rsqrtf(fmaf(cc, cc, ss * ss));
            c = cc * inv;
            s = ss * inv;
        }
        __syncthreads();

        // ---- build A operand
        v16bf af;
        #pragma unroll
        for (int e = 0; e < 8; ++e) { af[e] = a0.h[e]; af[8 + e] = a1.h[e]; }

        // ---- 4 N-subtiles: B fragment (col N = lane&15, 16 K / lane in table order) + WMMA
        #pragma unroll
        for (int s4 = 0; s4 < 4; ++s4) {
            int nb = jg * 64 + s4 * 16;
            v16bf bfv;
            #pragma unroll
            for (int e = 0; e < 16; ++e) {
                int kb = ((e & 8) << 1) + half * 8 + (e & 7);
                bfv[e] = Bs[kb * BSTR + nb + (lane & 15)];
            }
            acc[s4] = __builtin_amdgcn_wmma_f32_16x16x32_bf16(
                          false, af, false, bfv, (short)0, acc[s4], false, false);
        }
        __syncthreads();
    }

    // ---- store D: VGPR r -> M = half*8 + r within tile, N = lane&15 in subtile
    #pragma unroll
    for (int s4 = 0; s4 < 4; ++s4) {
        int n = n0 + jg * 64 + s4 * 16 + (lane & 15);
        if (n < T_OUT) {
            #pragma unroll
            for (int r = 0; r < 8; ++r) {
                int m = (mt << 4) + half * 8 + r;
                out[(size_t)n * BATCH + m] = acc[s4][r];
            }
        }
    }
}

// ---------------------------------------------------------------------------
__global__ void zero_scalar(float* p) { if (threadIdx.x == 0) p[0] = 0.0f; }

// ---------------------------------------------------------------------------
// Kernel 3: per-(band,batch) biquad cascade on both signals + fused Schroeder
// |suffix-sum| loss.  Pass 1: S = sum_t e[t], e = y_t^2 - y_a^2.
// Pass 2: L = sum_t |S - prefix_{<t}(e)|;  atomicAdd(L / (64*47040)).
// ---------------------------------------------------------------------------
__global__ __launch_bounds__(64)
void biquad_edc_loss(const float* __restrict__ sos,
                     const float* __restrict__ rir,
                     float* __restrict__ out_loss) {
    const int band = blockIdx.x;
    const int b    = threadIdx.x;

    const float* c = sos + band * NSEC * 6;   // all NSEC sections identical
    const float a0i = 1.0f / c[3];
    const float b0 = c[0] * a0i, b1 = c[1] * a0i, b2 = c[2] * a0i;
    const float a1 = c[4] * a0i, a2 = c[5] * a0i;

    const float* __restrict__ xt = rir;                      // [T][64]
    const float* __restrict__ xa = rir + (size_t)T_OUT * BATCH;

    float S = 0.0f, L = 0.0f;
    for (int pass = 0; pass < 2; ++pass) {
        float s1t[NSEC] = {}, s2t[NSEC] = {}, s1a[NSEC] = {}, s2a[NSEC] = {};
        float p = 0.0f, acc = 0.0f;
        for (int t = 0; t < T_OUT; ++t) {
            float vt = xt[(size_t)t * BATCH + b];
            float va = xa[(size_t)t * BATCH + b];
            __builtin_prefetch(&xt[(size_t)(t + 192) * BATCH + b], 0, 1);
            __builtin_prefetch(&xa[(size_t)(t + 192) * BATCH + b], 0, 1);
            #pragma unroll
            for (int j = 0; j < NSEC; ++j) {
                float yt = fmaf(b0, vt, s1t[j]);
                s1t[j] = fmaf(b1, vt, fmaf(-a1, yt, s2t[j]));
                s2t[j] = fmaf(b2, vt, -a2 * yt);
                vt = yt;
                float ya = fmaf(b0, va, s1a[j]);
                s1a[j] = fmaf(b1, va, fmaf(-a1, ya, s2a[j]));
                s2a[j] = fmaf(b2, va, -a2 * ya);
                va = ya;
            }
            float e = fmaf(vt, vt, -(va * va));
            if (pass == 0) {
                acc += e;
            } else {
                L += fabsf(S - p);
                p += e;
            }
        }
        if (pass == 0) S = acc;
    }
    atomicAdd(out_loss, L * (1.0f / (64.0f * 47040.0f)));
}

// ---------------------------------------------------------------------------
extern "C" void kernel_launch(void* const* d_in, const int* in_sizes, int n_in,
                              void* d_out, int out_size, void* d_ws, size_t ws_size,
                              hipStream_t stream) {
    const float* tre = (const float*)d_in[0];   // [64][48000]
    const float* tim = (const float*)d_in[1];
    const float* are = (const float*)d_in[2];
    const float* aim = (const float*)d_in[3];
    const float* sos = (const float*)d_in[4];   // [9][7][6]
    float* out = (float*)d_out;

    float*  rir  = (float*)d_ws;                              // 24.1 MB
    __bf16* apre = (__bf16*)((char*)d_ws +
                             (size_t)2 * T_OUT * BATCH * sizeof(float)); // 12.3 MB

    (void)in_sizes; (void)n_in; (void)out_size; (void)ws_size;

    // pack A (2*KTILES*2048 bf16 elements, one per thread)
    int packBlocks = (int)((2 * APRE_PER_SIG) / 256);         // exact: 24016
    pack_a_bf16<<<dim3(packBlocks), dim3(256), 0, stream>>>(tre, tim, are, aim, apre);

    dim3 gridA((T_OUT + NWG - 1) / NWG, 2);                   // 368 x 2 signals
    dft_irfft_wmma<<<gridA, dim3(256), 0, stream>>>(apre, rir);

    zero_scalar<<<dim3(1), dim3(64), 0, stream>>>(out);
    biquad_edc_loss<<<dim3(NBAND), dim3(BATCH), 0, stream>>>(sos, rir, out);
}
